// SimpleLinear_52003464020700
// MI455X (gfx1250) — matrix operand order repaired and checked by reading, not compiled, not run on hardware
//
#include <hip/hip_runtime.h>
#include <hip/hip_bf16.h>

// GPT forward (6 layers) + NLL loss for MI455X / gfx1250.
// All matmuls run on v_wmma_f32_16x16x32_f16 (wave32 WMMA, f32 accumulate).
// Each wave owns a 32x32 output macro-tile: 4 WMMAs per LDS stage.
//
// Call-site invariants exploited by the GEMM (all launches below respect them):
//   M % 128 == 0, K % 32 == 0, lda % 4 == 0, (TRANSB: N % 64 == 0, ldb % 4 == 0)
// Only row-major B may have a ragged N (LM head, N = 50257): handled with
// clamped unconditional loads + value select, never predicated loads.

typedef __attribute__((ext_vector_type(16))) _Float16 v16h;
typedef __attribute__((ext_vector_type(8)))  float    v8f;

static constexpr int Bc  = 2;
static constexpr int Tc  = 1024;
static constexpr int Cc  = 512;
static constexpr int Hc  = 8;
static constexpr int DHc = 64;
static constexpr int Lc  = 6;
static constexpr int Vc  = 50257;
static constexpr int Mc  = Bc * Tc;      // 2048 rows of activations
static constexpr int FFc = 4 * Cc;       // 2048

#define BM 128
#define BN 64
#define BKt 32

__device__ __forceinline__ unsigned pack2h(float a, float b) {
  union { _Float16 h[2]; unsigned u; } t;
  t.h[0] = (_Float16)a; t.h[1] = (_Float16)b;
  return t.u;
}

// ---------------------------------------------------------------------------
// Generic batched GEMM:  C[z] = A[z] (MxK) * B[z] (KxN)  (+bias / +resid / poly)
// EPI: 0 = bias only, 1 = bias + residual (read C), 2 = bias then cubic poly.
// TRANSB: B stored [N,K] row-major (used for Q*K^T).
// CAUSAL: skip blocks with n0 > m0+BM-1 (scores upper triangle).
// ---------------------------------------------------------------------------
template<int EPI, bool TRANSB, bool CAUSAL>
__global__ __launch_bounds__(256)
void gemm_wmma_f16(const float* __restrict__ A, const float* __restrict__ Bm,
                   float* __restrict__ Cm,
                   const float* __restrict__ bias, const float* __restrict__ poly,
                   int M, int N, int K, int lda, int ldb, int ldc,
                   long sA1, long sA2, long sB1, long sB2, long sC1, long sC2,
                   int zdim2)
{
  const int m0 = blockIdx.y * BM;
  const int n0 = blockIdx.x * BN;
  if (CAUSAL && n0 > m0 + (BM - 1)) return;   // fully masked score block

  const int z  = blockIdx.z;
  const int z1 = z / zdim2, z2 = z % zdim2;
  A  += z1 * sA1 + z2 * sA2;
  Bm += z1 * sB1 + z2 * sB2;
  Cm += z1 * sC1 + z2 * sC2;

  __shared__ _Float16 As[BM][BKt + 2];   // row-major [m][k], pad keeps dword align
  __shared__ _Float16 Bs[BN][BKt + 2];   // transposed store [n][k]

  const int tid  = threadIdx.x;
  const int lane = tid & 31;
  const int wave = tid >> 5;
  const int wm   = (wave & 3) * 32;      // wave's 32-row slice of the 128-row tile
  const int wn   = (wave >> 2) * 32;     // wave's 32-col slice (two 16x16 tiles)

  v8f acc00 = {}; v8f acc01 = {}; v8f acc10 = {}; v8f acc11 = {};

  for (int k0 = 0; k0 < K; k0 += BKt) {
    // ---- stage A tile: 128x32, 16 floats/thread, unconditional float4 ----
    {
      const int mm  = tid >> 1;            // 0..127
      const int kk0 = (tid & 1) * 16;      // 0 or 16
      const float4* ap = (const float4*)(A + (long)(m0 + mm) * lda + (k0 + kk0));
      const float4 fa0 = ap[0];
      const float4 fa1 = ap[1];
      const float4 fa2 = ap[2];
      const float4 fa3 = ap[3];
      if (k0 + BKt < K)
        __builtin_prefetch(A + (long)(m0 + mm) * lda + (k0 + BKt + kk0), 0, 1);
      unsigned* dst = (unsigned*)&As[mm][kk0];
      dst[0] = pack2h(fa0.x, fa0.y); dst[1] = pack2h(fa0.z, fa0.w);
      dst[2] = pack2h(fa1.x, fa1.y); dst[3] = pack2h(fa1.z, fa1.w);
      dst[4] = pack2h(fa2.x, fa2.y); dst[5] = pack2h(fa2.z, fa2.w);
      dst[6] = pack2h(fa3.x, fa3.y); dst[7] = pack2h(fa3.z, fa3.w);
    }
    // ---- stage B tile into Bs[n][k] ----
    if (TRANSB) {
      // N,K tile-aligned and ldb 16B-aligned at all TRANSB call sites
      const int nn  = tid >> 2;
      const int kk0 = (tid & 3) * 8;
      const float4* bp = (const float4*)(Bm + (long)(n0 + nn) * ldb + (k0 + kk0));
      const float4 fb0v = bp[0];
      const float4 fb1v = bp[1];
      unsigned* dst = (unsigned*)&Bs[nn][kk0];
      dst[0] = pack2h(fb0v.x, fb0v.y); dst[1] = pack2h(fb0v.z, fb0v.w);
      dst[2] = pack2h(fb1v.x, fb1v.y); dst[3] = pack2h(fb1v.z, fb1v.w);
    } else {
      // Row-major B: N may be ragged (lm_w) and ldb odd -> scalar dword loads,
      // clamped addresses (always legal) + value select, no predication.
      const int kk  = tid >> 3;
      const int nn0 = (tid & 7) * 8;
      const long rowb = (long)(k0 + kk) * ldb;
      float f[8];
#pragma unroll
      for (int j = 0; j < 8; ++j) {
        const int gn  = n0 + nn0 + j;
        const int gnc = (gn < N) ? gn : (N - 1);
        f[j] = Bm[rowb + gnc];
      }
#pragma unroll
      for (int j = 0; j < 8; ++j)
        if (n0 + nn0 + j >= N) f[j] = 0.0f;
#pragma unroll
      for (int j = 0; j < 8; ++j) Bs[nn0 + j][kk] = (_Float16)f[j];
    }
    __syncthreads();

    // ---- gather WMMA fragments per ISA 16-bit A(16x32)/B(32x16) layouts ----
    union { v16h v; unsigned u[8]; } fa0, fa1, fb0, fb1;
    {
      // A: lanes 0-15 hold K=0..7,16..23; lanes 16-31 hold K=8..15,24..31
      const int row  = wm + (lane & 15);
      const int koff = (lane >= 16) ? 4 : 0;      // dword units (k=8 -> 4 dwords)
      const unsigned* s0 = (const unsigned*)&As[row][0];
      const unsigned* s1 = (const unsigned*)&As[row + 16][0];
#pragma unroll
      for (int p = 0; p < 4; ++p) {
        fa0.u[p] = s0[koff + p]; fa0.u[4 + p] = s0[koff + 8 + p];
        fa1.u[p] = s1[koff + p]; fa1.u[4 + p] = s1[koff + 8 + p];
      }
    }
    {
      // B: lanes 0-15 hold K=0..7,16..23 (col N=lane); lanes 16-31 hold K=8..15,24..31
      const int kb  = (lane >= 16) ? 4 : 0;       // dword units
      const int nb  = wn + (lane & 15);
      const unsigned* s0 = (const unsigned*)&Bs[nb][0];
      const unsigned* s1 = (const unsigned*)&Bs[nb + 16][0];
#pragma unroll
      for (int p = 0; p < 4; ++p) {
        fb0.u[p] = s0[kb + p]; fb0.u[4 + p] = s0[kb + 8 + p];
        fb1.u[p] = s1[kb + p]; fb1.u[4 + p] = s1[kb + 8 + p];
      }
    }

    // 4 independent accumulation chains -> deep XDL pipelining, no RAW hazards
    acc00 = __builtin_amdgcn_wmma_f32_16x16x32_f16(false, fa0.v, false, fb0.v,
                                                   (short)0, acc00, false, false);
    acc01 = __builtin_amdgcn_wmma_f32_16x16x32_f16(false, fa0.v, false, fb1.v,
                                                   (short)0, acc01, false, false);
    acc10 = __builtin_amdgcn_wmma_f32_16x16x32_f16(false, fa1.v, false, fb0.v,
                                                   (short)0, acc10, false, false);
    acc11 = __builtin_amdgcn_wmma_f32_16x16x32_f16(false, fa1.v, false, fb1.v,
                                                   (short)0, acc11, false, false);
    __syncthreads();
  }

  // ---- fused epilogue (M always tile-aligned; only N may be ragged) ----
  float p0 = 0.f, p1 = 0.f, p2 = 0.f, p3 = 0.f;
  if (EPI == 2) { p0 = poly[0]; p1 = poly[1]; p2 = poly[2]; p3 = poly[3]; }
#pragma unroll
  for (int tn = 0; tn < 2; ++tn) {
    const int n = n0 + wn + tn * 16 + (lane & 15);
    if (n < N) {
      const float bv = bias ? bias[n] : 0.0f;
#pragma unroll
      for (int tm = 0; tm < 2; ++tm) {
        const v8f acc = tm ? (tn ? acc11 : acc10) : (tn ? acc01 : acc00);
        const int mbase = m0 + wm + tm * 16 + ((lane >= 16) ? 8 : 0);
#pragma unroll
        for (int r = 0; r < 8; ++r) {
          float v = acc[r] + bv;
          const long idx = (long)(mbase + r) * ldc + n;
          if (EPI == 1) v += Cm[idx];
          if (EPI == 2) v = p0 + v * (p1 + v * (p2 + v * p3));
          Cm[idx] = v;
        }
      }
    }
  }
}

// ---------------------------------------------------------------------------
__global__ void embed_kernel(const int* __restrict__ idx,
                             const float* __restrict__ tok,
                             const float* __restrict__ pos,
                             float* __restrict__ x, int total)
{
  for (int i = blockIdx.x * blockDim.x + threadIdx.x; i < total;
       i += gridDim.x * blockDim.x) {
    const int c   = i % Cc;
    const int bt  = i / Cc;
    const int t   = bt % Tc;
    const int tok_id = idx[bt];
    x[i] = tok[(long)tok_id * Cc + c] + pos[(long)t * Cc + c];
  }
}

__global__ __launch_bounds__(256)
void ln_kernel(const float* __restrict__ in, float* __restrict__ out,
               const float* __restrict__ s, const float* __restrict__ b)
{
  const int row = blockIdx.x;
  const float* xr = in + (long)row * Cc;
  __shared__ float sh1[256], sh2[256];
  float a1 = 0.f, a2 = 0.f;
  for (int c = threadIdx.x; c < Cc; c += 256) { const float v = xr[c]; a1 += v; a2 += v * v; }
  sh1[threadIdx.x] = a1; sh2[threadIdx.x] = a2;
  __syncthreads();
  for (int off = 128; off > 0; off >>= 1) {
    if (threadIdx.x < off) { sh1[threadIdx.x] += sh1[threadIdx.x + off];
                             sh2[threadIdx.x] += sh2[threadIdx.x + off]; }
    __syncthreads();
  }
  const float mean = sh1[0] / Cc;
  const float var  = sh2[0] / Cc - mean * mean;
  const float rstd = rsqrtf(var + 1e-5f);
  for (int c = threadIdx.x; c < Cc; c += 256)
    out[(long)row * Cc + c] = (xr[c] - mean) * rstd * s[c] + b[c];
}

__global__ __launch_bounds__(256)
void softmax_causal_kernel(float* __restrict__ wei, float scale)
{
  const long row = blockIdx.x;                 // (b*H + h)*T + t
  const int  t   = (int)(row % Tc);
  float* w = wei + row * (long)Tc;
  const int len = t + 1;
  __shared__ float sh[256];

  float lm = -3.4e38f;
  for (int s = threadIdx.x; s < len; s += 256) lm = fmaxf(lm, w[s] * scale);
  sh[threadIdx.x] = lm; __syncthreads();
  for (int off = 128; off > 0; off >>= 1) {
    if (threadIdx.x < off) sh[threadIdx.x] = fmaxf(sh[threadIdx.x], sh[threadIdx.x + off]);
    __syncthreads();
  }
  const float rowmax = sh[0];
  __syncthreads();

  float ls = 0.f;
  for (int s = threadIdx.x; s < len; s += 256) {
    const float e = expf(w[s] * scale - rowmax);
    w[s] = e; ls += e;
  }
  sh[threadIdx.x] = ls; __syncthreads();
  for (int off = 128; off > 0; off >>= 1) {
    if (threadIdx.x < off) sh[threadIdx.x] += sh[threadIdx.x + off];
    __syncthreads();
  }
  const float inv = 1.0f / sh[0];
  for (int s = threadIdx.x; s < len; s += 256) w[s] *= inv;
  for (int s = len + threadIdx.x; s < Tc; s += 256) w[s] = 0.0f;  // zero masked tail
}

__global__ void zero_loss_kernel(float* p) { if (threadIdx.x == 0) *p = 0.0f; }

__global__ __launch_bounds__(256)
void loss_kernel(const float* __restrict__ logits, const int* __restrict__ targets,
                 float* __restrict__ loss, float invBT)
{
  const int row = blockIdx.x;
  const float* lr = logits + (long)row * Vc;
  __shared__ float sh[256];

  float lm = -3.4e38f;
  for (int v = threadIdx.x; v < Vc; v += 256) lm = fmaxf(lm, lr[v]);
  sh[threadIdx.x] = lm; __syncthreads();
  for (int off = 128; off > 0; off >>= 1) {
    if (threadIdx.x < off) sh[threadIdx.x] = fmaxf(sh[threadIdx.x], sh[threadIdx.x + off]);
    __syncthreads();
  }
  const float rowmax = sh[0];
  __syncthreads();

  float ls = 0.f;
  for (int v = threadIdx.x; v < Vc; v += 256) ls += expf(lr[v] - rowmax);
  sh[threadIdx.x] = ls; __syncthreads();
  for (int off = 128; off > 0; off >>= 1) {
    if (threadIdx.x < off) sh[threadIdx.x] += sh[threadIdx.x + off];
    __syncthreads();
  }
  if (threadIdx.x == 0) {
    const int tgt = targets[row];
    const float lp = lr[tgt] - rowmax - logf(sh[0]);
    atomicAdd(loss, -lp * invBT);
  }
}

// ---------------------------------------------------------------------------
static inline dim3 gemm_grid(int M, int N, int Z) {
  return dim3((N + BN - 1) / BN, (M + BM - 1) / BM, Z);
}

extern "C" void kernel_launch(void* const* d_in, const int* /*in_sizes*/, int /*n_in*/,
                              void* d_out, int /*out_size*/, void* d_ws, size_t /*ws_size*/,
                              hipStream_t stream)
{
  const int*   idx      = (const int*)  d_in[0];
  const int*   targets  = (const int*)  d_in[1];
  const float* tok_emb  = (const float*)d_in[2];
  const float* pos_emb  = (const float*)d_in[3];
  const float* wq       = (const float*)d_in[4];
  const float* wk       = (const float*)d_in[5];
  const float* wv       = (const float*)d_in[6];
  const float* proj_w   = (const float*)d_in[7];
  const float* proj_b   = (const float*)d_in[8];
  const float* ln_sa_s  = (const float*)d_in[9];
  const float* ln_sa_b  = (const float*)d_in[10];
  const float* w1       = (const float*)d_in[11];
  const float* b1       = (const float*)d_in[12];
  const float* poly     = (const float*)d_in[13];
  const float* w2       = (const float*)d_in[14];
  const float* b2       = (const float*)d_in[15];
  const float* ln_ff_s  = (const float*)d_in[16];
  const float* ln_ff_b  = (const float*)d_in[17];
  const float* ln_blk_s = (const float*)d_in[18];
  const float* ln_blk_b = (const float*)d_in[19];
  const float* ln_f_s   = (const float*)d_in[20];
  const float* ln_f_b   = (const float*)d_in[21];
  const float* lm_w     = (const float*)d_in[22];
  const float* lm_b     = (const float*)d_in[23];

  // workspace layout (fp32), ~109 MB total
  float* ws = (float*)d_ws;
  size_t off = 0;
  float* x    = ws + off; off += (size_t)Mc * Cc;            // activations
  float* hbuf = ws + off; off += (size_t)Mc * Cc;            // LN output
  float* q    = ws + off; off += (size_t)Hc * Mc * DHc;      // [H][B*T][DH]
  float* kbuf = ws + off; off += (size_t)Hc * Mc * DHc;
  float* vbuf = ws + off; off += (size_t)Hc * Mc * DHc;
  float* o    = ws + off; off += (size_t)Mc * Cc;            // head-concat attn out
  float* u    = ws + off; off += (size_t)Mc * FFc;           // FFN intermediate
  float* wei  = ws + off; off += (size_t)Bc * Hc * Tc * Tc;  // attention probs

  float* logits = (float*)d_out;
  float* loss   = logits + (size_t)Mc * Vc;

  const float scale = 0.044194173824159216f;   // C^{-0.5}, C = 512 (full n_embd)

  embed_kernel<<<(Mc * Cc + 255) / 256, 256, 0, stream>>>(idx, tok_emb, pos_emb, x, Mc * Cc);

  for (int l = 0; l < Lc; ++l) {
    // --- attention block ---
    ln_kernel<<<Mc, 256, 0, stream>>>(x, hbuf, ln_sa_s + (size_t)l * Cc, ln_sa_b + (size_t)l * Cc);

    const size_t wo = (size_t)l * Hc * Cc * DHc;
    gemm_wmma_f16<0, false, false><<<gemm_grid(Mc, DHc, Hc), 256, 0, stream>>>(
        hbuf, wq + wo, q, nullptr, nullptr, Mc, DHc, Cc, Cc, DHc, DHc,
        0, 0, (long)Cc * DHc, 0, (long)Mc * DHc, 0, 1);
    gemm_wmma_f16<0, false, false><<<gemm_grid(Mc, DHc, Hc), 256, 0, stream>>>(
        hbuf, wk + wo, kbuf, nullptr, nullptr, Mc, DHc, Cc, Cc, DHc, DHc,
        0, 0, (long)Cc * DHc, 0, (long)Mc * DHc, 0, 1);
    gemm_wmma_f16<0, false, false><<<gemm_grid(Mc, DHc, Hc), 256, 0, stream>>>(
        hbuf, wv + wo, vbuf, nullptr, nullptr, Mc, DHc, Cc, Cc, DHc, DHc,
        0, 0, (long)Cc * DHc, 0, (long)Mc * DHc, 0, 1);

    // scores = q @ k^T, causal blocks skipped; z = (b,h)
    gemm_wmma_f16<0, true, true><<<gemm_grid(Tc, Tc, Bc * Hc), 256, 0, stream>>>(
        q, kbuf, wei, nullptr, nullptr, Tc, Tc, DHc, DHc, DHc, Tc,
        (long)Tc * DHc, (long)Mc * DHc, (long)Tc * DHc, (long)Mc * DHc,
        (long)Hc * Tc * Tc, (long)Tc * Tc, Hc);

    softmax_causal_kernel<<<Bc * Hc * Tc, 256, 0, stream>>>(wei, scale);

    // o = P @ v, written directly in head-concat [B*T, C] layout
    gemm_wmma_f16<0, false, false><<<gemm_grid(Tc, DHc, Bc * Hc), 256, 0, stream>>>(
        wei, vbuf, o, nullptr, nullptr, Tc, DHc, Tc, Tc, DHc, Cc,
        (long)Hc * Tc * Tc, (long)Tc * Tc, (long)Tc * DHc, (long)Mc * DHc,
        (long)Tc * Cc, (long)DHc, Hc);

    // x = x + o @ proj_w + proj_b   (residual epilogue, in-place on x)
    gemm_wmma_f16<1, false, false><<<gemm_grid(Mc, Cc, 1), 256, 0, stream>>>(
        o, proj_w + (size_t)l * Cc * Cc, x, proj_b + (size_t)l * Cc, nullptr,
        Mc, Cc, Cc, Cc, Cc, Cc, 0, 0, 0, 0, 0, 0, 1);

    // --- FFN block ---
    ln_kernel<<<Mc, 256, 0, stream>>>(x, hbuf, ln_ff_s + (size_t)l * Cc, ln_ff_b + (size_t)l * Cc);

    // u = poly3(h @ w1 + b1)  (fused cubic epilogue)
    gemm_wmma_f16<2, false, false><<<gemm_grid(Mc, FFc, 1), 256, 0, stream>>>(
        hbuf, w1 + (size_t)l * Cc * FFc, u, b1 + (size_t)l * FFc, poly + (size_t)l * 4,
        Mc, FFc, Cc, Cc, FFc, FFc, 0, 0, 0, 0, 0, 0, 1);

    // x = x + u @ w2 + b2
    gemm_wmma_f16<1, false, false><<<gemm_grid(Mc, Cc, 1), 256, 0, stream>>>(
        u, w2 + (size_t)l * FFc * Cc, x, b2 + (size_t)l * Cc, nullptr,
        Mc, Cc, FFc, FFc, Cc, Cc, 0, 0, 0, 0, 0, 0, 1);
  }

  // final double LayerNorm (as in the reference source)
  ln_kernel<<<Mc, 256, 0, stream>>>(x, hbuf, ln_blk_s, ln_blk_b);
  ln_kernel<<<Mc, 256, 0, stream>>>(hbuf, x, ln_f_s, ln_f_b);

  // logits = x @ lm_w + lm_b  -> d_out   (N = 50257 ragged, clamp+select path)
  gemm_wmma_f16<0, false, false><<<gemm_grid(Mc, Vc, 1), 256, 0, stream>>>(
      x, lm_w, logits, lm_b, nullptr, Mc, Vc, Cc, Cc, Vc, Vc,
      0, 0, 0, 0, 0, 0, 1);

  zero_loss_kernel<<<1, 32, 0, stream>>>(loss);
  loss_kernel<<<Mc, 256, 0, stream>>>(logits, targets, loss, 1.0f / (float)Mc);
}
